// TransformerDecoderConvModel_24653112279164
// MI455X (gfx1250) — matrix operand order repaired and checked by reading, not compiled, not run on hardware
//
#include <hip/hip_runtime.h>
#include <hip/hip_bf16.h>
#include <cstdint>

// ---------------------------------------------------------------------------
// CDNA5 (gfx1250) implementation of the conv-attention transformer decoder.
// All GEMM-shaped work runs through v_wmma_f32_16x16x32_bf16 (wave32 WMMA).
// GEMM: 128x128 block tile, 8 waves x (32x64), double-buffered LDS pipeline.
// ---------------------------------------------------------------------------

typedef __attribute__((ext_vector_type(16))) __bf16 v16bf;
typedef __attribute__((ext_vector_type(8)))  float  v8f;

union BFrag { v16bf v; uint32_t u[8]; };

__device__ __forceinline__ uint16_t f2bf(float f) {
  uint32_t u = __float_as_uint(f);
  uint32_t r = (u + 0x7FFFu + ((u >> 16) & 1u)) >> 16;  // round-to-nearest-even
  return (uint16_t)r;
}

__device__ __forceinline__ v8f vzero8() {
  v8f z = {0.f, 0.f, 0.f, 0.f, 0.f, 0.f, 0.f, 0.f};
  return z;
}

// ---------------------------------------------------------------------------
// Generic bf16 WMMA GEMM:  C[M,N] = act( sum_taps A_shifted[M,K] * Bt[N,K]^T + bias )
// Bt row-major [N,K]. taps==3 implements the causal conv (tap j shifted back
// by 2-j rows, zero-padded at each sequence start).
// Software pipeline: global loads for tile t+1 are issued before the WMMA
// block of tile t; LDS is double buffered; one barrier per iteration.
// ---------------------------------------------------------------------------
__global__ __launch_bounds__(256) void k_gemm(
    const uint16_t* __restrict__ A, int M, int K,
    const uint16_t* __restrict__ Bt, int N, int taps, int tapStride,
    int seqLen,
    const float* __restrict__ bias,
    float* __restrict__ Cf, uint16_t* __restrict__ Cb,
    int relu)
{
  __shared__ __align__(16) uint32_t sA[2][128 * 16];  // 128 rows x 32 bf16
  __shared__ __align__(16) uint32_t sB[2][128 * 16];
  const int tid  = threadIdx.x;
  const int lane = tid & 31, wave = tid >> 5;
  const int wm = wave & 3, wn = wave >> 2;       // 4 M-waves x 2 N-waves
  const int half = lane >> 4, l16 = lane & 15;
  const int lo4 = half * 4;
  const int blockM = blockIdx.y * 128, blockN = blockIdx.x * 128;
  const int crow = tid >> 2;                     // staging chunk row (0..63)
  const int coff = (tid & 3) << 2;               // staging uint offset
  const int kTiles = K >> 5;
  const int nIter = taps * kTiles;

  __builtin_amdgcn_s_wait_tensorcnt(0);

  auto loadTiles = [&](int it, uint4* ra, uint4* rb) {
    int tp = (taps == 3) ? (it / kTiles) : 0;
    int k0 = (it - tp * kTiles) << 5;
    int shift = (taps == 3) ? (2 - tp) : 0;
    const uint16_t* Bp = Bt + (size_t)tp * tapStride;
#pragma unroll
    for (int c = 0; c < 2; ++c) {
      int r = blockM + crow + c * 64;
      uint4 va = make_uint4(0u, 0u, 0u, 0u);
      if (r < M && ((r & (seqLen - 1)) >= shift)) {
        const uint32_t* s = (const uint32_t*)(A + (size_t)(r - shift) * K + k0);
        va = *(const uint4*)(s + coff);
        if (k0 + 32 < K) __builtin_prefetch(s + coff + 16, 0, 1);
      }
      ra[c] = va;
      int n = blockN + crow + c * 64;
      uint4 vb = make_uint4(0u, 0u, 0u, 0u);
      if (n < N) {
        const uint32_t* s = (const uint32_t*)(Bp + (size_t)n * K + k0);
        vb = *(const uint4*)(s + coff);
        if (k0 + 32 < K) __builtin_prefetch(s + coff + 16, 0, 1);
      }
      rb[c] = vb;
    }
  };
  auto storeTiles = [&](int buf, const uint4* ra, const uint4* rb) {
#pragma unroll
    for (int c = 0; c < 2; ++c) {
      *(uint4*)&sA[buf][(crow + c * 64) * 16 + coff] = ra[c];
      *(uint4*)&sB[buf][(crow + c * 64) * 16 + coff] = rb[c];
    }
  };

  v8f acc[2][4];
#pragma unroll
  for (int mi = 0; mi < 2; ++mi)
#pragma unroll
    for (int ni = 0; ni < 4; ++ni) acc[mi][ni] = vzero8();

  uint4 ra[2], rb[2];
  loadTiles(0, ra, rb);
  storeTiles(0, ra, rb);
  int cur = 0;

  for (int it = 0; it < nIter; ++it) {
    __syncthreads();
    const bool more = (it + 1 < nIter);
    if (more) loadTiles(it + 1, ra, rb);   // overlap with WMMA below

    const uint32_t* baseA = &sA[cur][0];
    const uint32_t* baseB = &sB[cur][0];
    BFrag a[2], b[4];
#pragma unroll
    for (int mi = 0; mi < 2; ++mi) {
      const uint32_t* pr = baseA + (wm * 32 + mi * 16 + l16) * 16;
#pragma unroll
      for (int j = 0; j < 4; ++j) {
        a[mi].u[j]     = pr[lo4 + j];       // K = lo..lo+7
        a[mi].u[4 + j] = pr[8 + lo4 + j];   // K = 16+lo..23+lo
      }
    }
#pragma unroll
    for (int ni = 0; ni < 4; ++ni) {
      const uint32_t* pr = baseB + (wn * 64 + ni * 16 + l16) * 16 + half * 4;
#pragma unroll
      for (int j = 0; j < 8; ++j) b[ni].u[j] = pr[(j & 3) + ((j >> 2) << 3)];  // K = lo..lo+7, 16+lo..23+lo
    }
#pragma unroll
    for (int mi = 0; mi < 2; ++mi)
#pragma unroll
      for (int ni = 0; ni < 4; ++ni)
        acc[mi][ni] = __builtin_amdgcn_wmma_f32_16x16x32_bf16(
            false, a[mi].v, false, b[ni].v, (short)0, acc[mi][ni],
            false, false);

    if (more) {
      storeTiles(1 - cur, ra, rb);
      cur ^= 1;
    }
  }

  // ---- epilogue ----
#pragma unroll
  for (int mi = 0; mi < 2; ++mi)
#pragma unroll
    for (int ni = 0; ni < 4; ++ni)
#pragma unroll
      for (int i = 0; i < 8; ++i) {
        int rg = blockM + wm * 32 + mi * 16 + i + half * 8;
        int cg = blockN + wn * 64 + ni * 16 + l16;
        if (rg < M && cg < N) {
          float v = acc[mi][ni][i];
          if (bias) v += bias[cg];
          if (relu) v = fmaxf(v, 0.0f);
          size_t o = (size_t)rg * N + cg;
          if (Cf) Cf[o] = v;
          if (Cb) Cb[o] = f2bf(v);
        }
      }
}

// ---------------------------------------------------------------------------
// Fused attention for one (b, h, 16-query-row tile): scores = QK^T/8 (+mask),
// softmax in LDS (all 32 lanes: 2 segments/row + shfl_xor combine), then
// P * V with WMMA. One wave per workgroup.
// Qb/Kb: [B*S, 512] bf16; Vt: [B*H, 64, S] bf16 (pre-transposed).
// mask!=0 keeps strictly-upper positions (kc > qr), rest -> -1e9 (faithful
// to the reference's buggy masked_fill).
// ---------------------------------------------------------------------------
__global__ __launch_bounds__(32) void k_attn(
    const uint16_t* __restrict__ Qb, const uint16_t* __restrict__ Kb,
    const uint16_t* __restrict__ Vt, uint16_t* __restrict__ Ob,
    int S, int mask)
{
  __shared__ __align__(16) float    sc[16][512];
  __shared__ __align__(16) uint16_t pb[16][512];
  const int lane = threadIdx.x;
  const int half = lane >> 4, l16 = lane & 15;
  const int lo4 = half * 4;
  const int qt = blockIdx.x * 16;
  const int bh = blockIdx.y, b = bh >> 3, h = bh & 7;
  const size_t tokBase = (size_t)b * S;

  // Q fragment: rows qt..qt+15, dk=64 -> two 16x32 A fragments
  BFrag aq[2];
  {
    const uint32_t* qp =
        (const uint32_t*)(Qb + (tokBase + qt + l16) * 512 + h * 64);
#pragma unroll
    for (int c = 0; c < 2; ++c)
#pragma unroll
      for (int j = 0; j < 4; ++j) {
        aq[c].u[j]     = qp[c * 16 + lo4 + j];
        aq[c].u[4 + j] = qp[c * 16 + 8 + lo4 + j];
      }
  }
  // ---- scores ----
  for (int kb = 0; kb < S; kb += 16) {
    v8f acc = vzero8();
    const uint32_t* kp =
        (const uint32_t*)(Kb + (tokBase + kb + l16) * 512 + h * 64);
#pragma unroll
    for (int c = 0; c < 2; ++c) {
      BFrag bk;
#pragma unroll
      for (int j = 0; j < 8; ++j) bk.u[j] = kp[c * 16 + half * 4 + (j & 3) + ((j >> 2) << 3)];
      acc = __builtin_amdgcn_wmma_f32_16x16x32_bf16(
          false, aq[c].v, false, bk.v, (short)0, acc, false, false);
    }
#pragma unroll
    for (int i = 0; i < 8; ++i) {
      int qr = qt + i + half * 8;
      int kc = kb + l16;
      float v = acc[i] * 0.125f;               // 1/sqrt(64)
      if (mask && kc <= qr) v = -1e9f;         // keep strictly-upper only
      sc[i + half * 8][kc] = v;
    }
  }
  __syncthreads();
  // ---- softmax: row = l16, segment = half (256 cols each) ----
  {
    float* row = &sc[l16][half * 256];
    float mx = -3.4e38f;
    for (int j = 0; j < 256; ++j) mx = fmaxf(mx, row[j]);
    mx = fmaxf(mx, __shfl_xor(mx, 16, 32));
    float sum = 0.0f;
    for (int j = 0; j < 256; ++j) {
      float e = __expf(row[j] - mx);
      row[j] = e;
      sum += e;
    }
    sum += __shfl_xor(sum, 16, 32);
    float inv = 1.0f / sum;
    uint16_t* prow = &pb[l16][half * 256];
    for (int j = 0; j < 256; ++j) prow[j] = f2bf(row[j] * inv);
  }
  __syncthreads();
  // ---- P @ V : out tile 16 x 64 ----
  v8f o[4];
  o[0] = vzero8(); o[1] = vzero8(); o[2] = vzero8(); o[3] = vzero8();
  for (int k0 = 0; k0 < S; k0 += 32) {
    BFrag ap;
    const uint32_t* pr = (const uint32_t*)&pb[l16][0];
#pragma unroll
    for (int j = 0; j < 4; ++j) {
      ap.u[j]     = pr[k0 / 2 + lo4 + j];
      ap.u[4 + j] = pr[k0 / 2 + 8 + lo4 + j];
    }
#pragma unroll
    for (int ni = 0; ni < 4; ++ni) {
      const uint32_t* vp = (const uint32_t*)(
          Vt + ((size_t)bh * 64 + ni * 16 + l16) * 512 + k0);
      BFrag bv;
#pragma unroll
      for (int j = 0; j < 8; ++j) bv.u[j] = vp[half * 4 + (j & 3) + ((j >> 2) << 3)];
      o[ni] = __builtin_amdgcn_wmma_f32_16x16x32_bf16(
          false, ap.v, false, bv.v, (short)0, o[ni], false, false);
    }
  }
#pragma unroll
  for (int ni = 0; ni < 4; ++ni)
#pragma unroll
    for (int i = 0; i < 8; ++i) {
      size_t r = tokBase + qt + i + half * 8;
      int col = h * 64 + ni * 16 + l16;
      Ob[r * 512 + col] = f2bf(o[ni][i]);
    }
}

// ---------------------------------------------------------------------------
// Embedding: out[r,:] = in[r,0:8] @ W[512,8]^T + bias + scale * PE(t)
// ---------------------------------------------------------------------------
__global__ __launch_bounds__(256) void k_embed(
    const float* __restrict__ in, const float* __restrict__ W,
    const float* __restrict__ bias, const float* __restrict__ scale,
    int S, float* __restrict__ Of, uint16_t* __restrict__ Ob)
{
  const int r = blockIdx.x;
  const int t = r & (S - 1);
  const float sc = scale[0];
  float xin[8];
#pragma unroll
  for (int j = 0; j < 8; ++j) xin[j] = in[(size_t)r * 8 + j];
  for (int d = threadIdx.x; d < 512; d += 256) {
    float acc = bias[d];
#pragma unroll
    for (int j = 0; j < 8; ++j) acc += xin[j] * W[d * 8 + j];
    float div = __expf(-(float)(d & ~1) * 0.0179889460f);  // ln(1e4)/512
    float ang = (float)t * div;
    acc += sc * ((d & 1) ? __cosf(ang) : __sinf(ang));
    Of[(size_t)r * 512 + d] = acc;
    Ob[(size_t)r * 512 + d] = f2bf(acc);
  }
}

// fp32 -> bf16 weight convert with optional conv-tap de-interleave:
// dst[i] = bf16( src[i*ntaps + tap] )
__global__ __launch_bounds__(256) void k_conv_w(
    const float* __restrict__ src, uint16_t* __restrict__ dst,
    int n, int ntaps, int tap)
{
  int i = blockIdx.x * 256 + threadIdx.x;
  if (i < n) dst[i] = f2bf(src[(size_t)i * ntaps + tap]);
}

// V transpose: Vt[bh][d][k] = Vb[(b*S+k)*512 + h*64 + d]
__global__ __launch_bounds__(256) void k_trans_v(
    const uint16_t* __restrict__ Vb, uint16_t* __restrict__ Vt, int S)
{
  size_t i = (size_t)blockIdx.x * 256 + threadIdx.x;  // over B*H*64*S
  int k = (int)(i & (size_t)(S - 1));
  size_t rem = i / (size_t)S;
  int d = (int)(rem & 63);
  size_t bh = rem >> 6;
  int b = (int)(bh >> 3), h = (int)(bh & 7);
  Vt[i] = Vb[((size_t)(b * S + k)) * 512 + h * 64 + d];
}

// Fused residual + LayerNorm over 512 columns (one row per block).
__global__ __launch_bounds__(256) void k_add_ln(
    const float* __restrict__ X, const float* __restrict__ Sub,
    const float* __restrict__ g, const float* __restrict__ bia,
    float* __restrict__ Of, uint16_t* __restrict__ Ob)
{
  __shared__ float red[256];
  const int row = blockIdx.x, tid = threadIdx.x;
  const float* xr = X + (size_t)row * 512;
  const float* sr = Sub + (size_t)row * 512;
  float a  = xr[tid] + sr[tid];
  float b2 = xr[tid + 256] + sr[tid + 256];
  red[tid] = a + b2;
  __syncthreads();
  for (int s = 128; s > 0; s >>= 1) {
    if (tid < s) red[tid] += red[tid + s];
    __syncthreads();
  }
  float mean = red[0] * (1.0f / 512.0f);
  __syncthreads();
  float da = a - mean, db = b2 - mean;
  red[tid] = da * da + db * db;
  __syncthreads();
  for (int s = 128; s > 0; s >>= 1) {
    if (tid < s) red[tid] += red[tid + s];
    __syncthreads();
  }
  float rstd = rsqrtf(red[0] * (1.0f / 512.0f) + 1e-5f);
  float o1 = da * rstd * g[tid] + bia[tid];
  float o2 = db * rstd * g[tid + 256] + bia[tid + 256];
  float* orow = Of + (size_t)row * 512;
  uint16_t* brow = Ob + (size_t)row * 512;
  orow[tid] = o1;           orow[tid + 256] = o2;
  brow[tid] = f2bf(o1);     brow[tid + 256] = f2bf(o2);
}

// ---------------------------------------------------------------------------
// Host orchestration
// ---------------------------------------------------------------------------
extern "C" void kernel_launch(void* const* d_in, const int* in_sizes, int n_in,
                              void* d_out, int out_size, void* d_ws, size_t ws_size,
                              hipStream_t stream) {
  (void)in_sizes; (void)out_size; (void)ws_size;
  const int S = 512, Bsz = 16, H = 8, TOK = Bsz * S;
  if (n_in < 88) return;

  // jax flattens the params dict by sorted keys:
  //  0 src, 1 trg, 2 dec_b, 3 dec_w, 4 enc_b, 5 enc_w, 6 fc_b, 7 fc_w,
  //  8.. 3 layers x 26 sorted leaves, 86 scale_dec, 87 scale_enc
  const float* src = (const float*)d_in[0];
  const float* trg = (const float*)d_in[1];
  const float* dec_b = (const float*)d_in[2];
  const float* dec_w = (const float*)d_in[3];
  const float* enc_b = (const float*)d_in[4];
  const float* enc_w = (const float*)d_in[5];
  const float* fc_b  = (const float*)d_in[6];
  const float* fc_w  = (const float*)d_in[7];
  const float* scale_dec = (const float*)d_in[86];
  const float* scale_enc = (const float*)d_in[87];
  enum { CA_K_B = 0, CA_K_W, CA_Q_B, CA_Q_W, CA_V_B, CA_V_W, CA_O_B, CA_O_W,
         FF1_B, FF1_W, FF2_B, FF2_W, LN1_B, LN1_G, LN2_B, LN2_G, LN3_B, LN3_G,
         SA_K_B, SA_K_W, SA_Q_B, SA_Q_W, SA_V_B, SA_V_W, SA_O_B, SA_O_W };
  auto LPf = [&](int l, int idx) -> const float* {
    return (const float*)d_in[8 + l * 26 + idx];
  };

  // ---- workspace carve-up ----
  char* ws = (char*)d_ws;
  size_t off = 0;
  auto alloc = [&](size_t bytes) -> void* {
    void* p = ws + off;
    off = (off + bytes + 255) & ~(size_t)255;
    return p;
  };
  float*    mem_f = (float*)alloc((size_t)TOK * 512 * 4);
  uint16_t* mem_b = (uint16_t*)alloc((size_t)TOK * 512 * 2);
  float*    x_f   = (float*)alloc((size_t)TOK * 512 * 4);
  uint16_t* x_b   = (uint16_t*)alloc((size_t)TOK * 512 * 2);
  uint16_t* q_b   = (uint16_t*)alloc((size_t)TOK * 512 * 2);
  uint16_t* k_b   = (uint16_t*)alloc((size_t)TOK * 512 * 2);
  uint16_t* v_b   = (uint16_t*)alloc((size_t)TOK * 512 * 2);
  uint16_t* vt_b  = (uint16_t*)alloc((size_t)TOK * 512 * 2);
  uint16_t* at_b  = (uint16_t*)alloc((size_t)TOK * 512 * 2);
  float*    pr_f  = (float*)alloc((size_t)TOK * 512 * 4);
  uint16_t* ffh_b = (uint16_t*)alloc((size_t)TOK * 2048 * 2);
  uint16_t* w_q   = (uint16_t*)alloc((size_t)3 * 262144 * 2);
  uint16_t* w_k   = (uint16_t*)alloc((size_t)3 * 262144 * 2);
  uint16_t* w_v   = (uint16_t*)alloc((size_t)262144 * 2);
  uint16_t* w_o   = (uint16_t*)alloc((size_t)262144 * 2);
  uint16_t* w_f1  = (uint16_t*)alloc((size_t)1048576 * 2);
  uint16_t* w_f2  = (uint16_t*)alloc((size_t)1048576 * 2);
  uint16_t* w_fc  = (uint16_t*)alloc((size_t)4096 * 2);

  const dim3 blk256(256);
  auto gemm = [&](const uint16_t* A, int M, int K, const uint16_t* Bt, int N,
                  int taps, int tapStride, const float* bias,
                  float* Cf, uint16_t* Cb, int relu) {
    dim3 grid((N + 127) / 128, (M + 127) / 128);
    k_gemm<<<grid, blk256, 0, stream>>>(A, M, K, Bt, N, taps, tapStride, S,
                                        bias, Cf, Cb, relu);
  };
  auto convw = [&](const float* s, uint16_t* dst, int n, int ntaps, int tap) {
    k_conv_w<<<dim3((n + 255) / 256), blk256, 0, stream>>>(s, dst, n, ntaps, tap);
  };

  // ---- embeddings ----
  k_embed<<<dim3(TOK), blk256, 0, stream>>>(src, enc_w, enc_b, scale_enc, S, mem_f, mem_b);
  k_embed<<<dim3(TOK), blk256, 0, stream>>>(trg, dec_w, dec_b, scale_dec, S, x_f, x_b);

  for (int l = 0; l < 3; ++l) {
    // ======== self attention ========
    for (int t = 0; t < 3; ++t) {
      convw(LPf(l, SA_Q_W), w_q + (size_t)t * 262144, 262144, 3, t);
      convw(LPf(l, SA_K_W), w_k + (size_t)t * 262144, 262144, 3, t);
    }
    convw(LPf(l, SA_V_W), w_v, 262144, 1, 0);
    convw(LPf(l, SA_O_W), w_o, 262144, 1, 0);
    gemm(x_b, TOK, 512, w_q, 512, 3, 262144, LPf(l, SA_Q_B), nullptr, q_b, 0);
    gemm(x_b, TOK, 512, w_k, 512, 3, 262144, LPf(l, SA_K_B), nullptr, k_b, 0);
    gemm(x_b, TOK, 512, w_v, 512, 1, 0,      LPf(l, SA_V_B), nullptr, v_b, 0);
    k_trans_v<<<dim3((TOK * 512) / 256), blk256, 0, stream>>>(v_b, vt_b, S);
    k_attn<<<dim3(S / 16, Bsz * H), dim3(32), 0, stream>>>(q_b, k_b, vt_b, at_b, S, 1);
    gemm(at_b, TOK, 512, w_o, 512, 1, 0, LPf(l, SA_O_B), pr_f, nullptr, 0);
    k_add_ln<<<dim3(TOK), blk256, 0, stream>>>(x_f, pr_f, LPf(l, LN1_G), LPf(l, LN1_B), x_f, x_b);

    // ======== cross attention ========
    for (int t = 0; t < 3; ++t) {
      convw(LPf(l, CA_Q_W), w_q + (size_t)t * 262144, 262144, 3, t);
      convw(LPf(l, CA_K_W), w_k + (size_t)t * 262144, 262144, 3, t);
    }
    convw(LPf(l, CA_V_W), w_v, 262144, 1, 0);
    convw(LPf(l, CA_O_W), w_o, 262144, 1, 0);
    gemm(x_b,   TOK, 512, w_q, 512, 3, 262144, LPf(l, CA_Q_B), nullptr, q_b, 0);
    gemm(mem_b, TOK, 512, w_k, 512, 3, 262144, LPf(l, CA_K_B), nullptr, k_b, 0);
    gemm(mem_b, TOK, 512, w_v, 512, 1, 0,      LPf(l, CA_V_B), nullptr, v_b, 0);
    k_trans_v<<<dim3((TOK * 512) / 256), blk256, 0, stream>>>(v_b, vt_b, S);
    k_attn<<<dim3(S / 16, Bsz * H), dim3(32), 0, stream>>>(q_b, k_b, vt_b, at_b, S, 0);
    gemm(at_b, TOK, 512, w_o, 512, 1, 0, LPf(l, CA_O_B), pr_f, nullptr, 0);
    k_add_ln<<<dim3(TOK), blk256, 0, stream>>>(x_f, pr_f, LPf(l, LN2_G), LPf(l, LN2_B), x_f, x_b);

    // ======== feed-forward ========
    convw(LPf(l, FF1_W), w_f1, 1048576, 1, 0);
    convw(LPf(l, FF2_W), w_f2, 1048576, 1, 0);
    gemm(x_b,   TOK, 512,  w_f1, 2048, 1, 0, LPf(l, FF1_B), nullptr, ffh_b, 1);
    gemm(ffh_b, TOK, 2048, w_f2, 512,  1, 0, LPf(l, FF2_B), pr_f, nullptr, 0);
    k_add_ln<<<dim3(TOK), blk256, 0, stream>>>(x_f, pr_f, LPf(l, LN3_G), LPf(l, LN3_B), x_f, x_b);
  }

  // ---- final projection to (B, S, 8) ----
  convw(fc_w, w_fc, 4096, 1, 0);
  gemm(x_b, TOK, 512, w_fc, 8, 1, 0, fc_b, (float*)d_out, nullptr, 0);
}
